// FP8GraphableDecodeStep_87376814669975
// MI455X (gfx1250) — 8-bit matrix operand order repaired and checked by reading, compiled, not run on hardware
//
#include <hip/hip_runtime.h>

#define N_LAYERS 4
#define D        2048
#define H        32
#define KVH      8
#define HD       64
#define REP      4
#define INTER    8192
#define VOCAB    32000
#define MAX_SEQ  4096
#define CACHE_LEN 2047
#define CUR_LEN   2048
#define KCHUNK   512

typedef int   v16i __attribute__((ext_vector_type(16)));
typedef float v8f  __attribute__((ext_vector_type(8)));

__device__ inline unsigned int f32_to_e4m3_sw(float x) {
  unsigned int u    = __float_as_uint(x);
  unsigned int sign = (u >> 24) & 0x80u;
  float ax = fabsf(x);
  if (!(ax == ax)) return sign | 0x7Fu;
  if (ax > 448.0f) return sign | 0x7Eu;
  if (ax < 0.0009765625f) return sign;
  int e; float m = frexpf(ax, &e);
  int ue = e - 1;
  if (ue < -6) {
    unsigned int n = (unsigned int)(ax * 512.0f + 0.5f);
    if (n >= 8u) return sign | 0x08u;
    return sign | n;
  }
  unsigned int n = (unsigned int)(m * 16.0f + 0.5f);
  if (n >= 16u) { n = 8u; ue += 1; if (ue > 8) return sign | 0x7Eu; }
  return sign | ((unsigned int)(ue + 7) << 3) | (n & 7u);
}

__device__ inline unsigned int pk4_e4m3(float a, float b, float c, float d) {
#if __has_builtin(__builtin_amdgcn_cvt_pk_fp8_f32)
  int lo = __builtin_amdgcn_cvt_pk_fp8_f32(a, b, 0, false);
  int hi = __builtin_amdgcn_cvt_pk_fp8_f32(c, d, lo, true);
  return (unsigned int)hi;
#else
  return  f32_to_e4m3_sw(a)        | (f32_to_e4m3_sw(b) << 8) |
         (f32_to_e4m3_sw(c) << 16) | (f32_to_e4m3_sw(d) << 24);
#endif
}

__device__ inline float silu_f(float x) { return x / (1.0f + __expf(-x)); }

__global__ void zero_amax_kernel(unsigned int* amax) {
  if (threadIdx.x < 28) amax[threadIdx.x] = 0u;
}

__global__ void __launch_bounds__(256)
amax_kernel(const float* __restrict__ w, int n4, unsigned int* __restrict__ amax) {
  __shared__ float red[256];
  int tid = threadIdx.x;
  float m = 0.0f;
  for (int i = blockIdx.x * blockDim.x + tid; i < n4; i += gridDim.x * blockDim.x) {
    float4 v = ((const float4*)w)[i];
    m = fmaxf(m, fmaxf(fmaxf(fabsf(v.x), fabsf(v.y)), fmaxf(fabsf(v.z), fabsf(v.w))));
  }
  red[tid] = m; __syncthreads();
  for (int s = 128; s > 0; s >>= 1) { if (tid < s) red[tid] = fmaxf(red[tid], red[tid + s]); __syncthreads(); }
  if (tid == 0) atomicMax(amax, __float_as_uint(red[0]));
}

__global__ void finalize_scales_kernel(const unsigned int* amax, float* scales) {
  int i = threadIdx.x;
  if (i < 28) {
    float a = __uint_as_float(amax[i]);
    scales[i] = (a > 0.0f) ? (a / 448.0f) : 1.0f;
  }
}

__global__ void embed_kernel(const float* __restrict__ emb, const int* __restrict__ ids,
                             float* __restrict__ h) {
  int i = blockIdx.x * blockDim.x + threadIdx.x;
  if (i < D) h[i] = emb[(size_t)ids[0] * D + i];
}

__global__ void add_kernel(float* __restrict__ h, const float* __restrict__ a, int n) {
  int i = blockIdx.x * blockDim.x + threadIdx.x;
  if (i < n) h[i] += a[i];
}

__global__ void quant_kernel(const float* __restrict__ x, unsigned int* __restrict__ xq, int n4) {
  int i = blockIdx.x * blockDim.x + threadIdx.x;
  if (i < n4) {
    float4 v = ((const float4*)x)[i];
    xq[i] = pk4_e4m3(v.x, v.y, v.z, v.w);
  }
}

__global__ void silu_mul_q_kernel(const float* __restrict__ g, const float* __restrict__ u,
                                  unsigned int* __restrict__ xq, int n4) {
  int i = blockIdx.x * blockDim.x + threadIdx.x;
  if (i < n4) {
    float4 gv = ((const float4*)g)[i];
    float4 uv = ((const float4*)u)[i];
    xq[i] = pk4_e4m3(silu_f(gv.x) * uv.x, silu_f(gv.y) * uv.y,
                     silu_f(gv.z) * uv.z, silu_f(gv.w) * uv.w);
  }
}

__global__ void reduce_kernel(const float* __restrict__ part, float* __restrict__ out,
                              int N, int nsplit) {
  int n = blockIdx.x * blockDim.x + threadIdx.x;
  if (n < N) {
    float s = 0.0f;
    for (int i = 0; i < nsplit; ++i) s += part[(size_t)i * N + n];
    out[n] = s;
  }
}

__global__ void __launch_bounds__(256)
rmsnorm_kernel(const float* __restrict__ x, const float* __restrict__ w,
               float* __restrict__ yout, unsigned int* __restrict__ yq, int n) {
  __shared__ float red[256];
  int tid = threadIdx.x;
  float s = 0.0f;
  for (int i = tid; i < n; i += 256) { float v = x[i]; s += v * v; }
  red[tid] = s; __syncthreads();
  for (int st = 128; st > 0; st >>= 1) { if (tid < st) red[tid] += red[tid + st]; __syncthreads(); }
  float inv = rsqrtf(red[0] * (1.0f / n) + 1e-6f);
  for (int i = tid; i * 4 < n; i += 256) {
    float4 v  = ((const float4*)x)[i];
    float4 wv = ((const float4*)w)[i];
    float a = v.x * inv * wv.x, b = v.y * inv * wv.y;
    float c = v.z * inv * wv.z, d = v.w * inv * wv.w;
    if (yout) ((float4*)yout)[i] = make_float4(a, b, c, d);
    if (yq)   yq[i] = pk4_e4m3(a, b, c, d);
  }
}

__global__ void rope_kernel(const float* __restrict__ q, const float* __restrict__ k,
                            const int* __restrict__ pos, float* __restrict__ qr,
                            float* __restrict__ kr) {
  int i = blockIdx.x * blockDim.x + threadIdx.x;
  if (i >= (H + KVH) * HD) return;
  int h = i >> 6, d = i & 63;
  const float* src; float* dst;
  if (h < H) { src = q + h * HD; dst = qr + h * HD; }
  else       { src = k + (h - H) * HD; dst = kr + (h - H) * HD; }
  int j = d & 31;
  float freq = __powf(10000.0f, -2.0f * (float)j / (float)HD);
  float ang  = (float)pos[0] * freq;
  float c = __cosf(ang), s = __sinf(ang);
  float x = src[d];
  float other = (d < 32) ? -src[d + 32] : src[d - 32];
  dst[d] = x * c + other * s;
}

__global__ void __launch_bounds__(32)
gemv_fp8_wmma_kernel(const float* __restrict__ W, const unsigned int* __restrict__ xq,
                     const float* __restrict__ scale_ptr, float* __restrict__ part,
                     int K, int N) {
  const int lane  = threadIdx.x;
  const int m     = lane & 15;
  const int half  = lane >> 4;
  const int n0    = blockIdx.x * 16;
  const int split = blockIdx.y;
  const int k0    = split * KCHUNK;
  const float s     = scale_ptr[0];
  const float inv_s = 1.0f / s;
  const float* wr = W + (size_t)(n0 + m) * (size_t)K;
  const unsigned char* xb = (const unsigned char*)xq;
  v8f acc = {0.f, 0.f, 0.f, 0.f, 0.f, 0.f, 0.f, 0.f};
  for (int kb = k0; kb < k0 + KCHUNK; kb += 128) {
    v16i A;
#pragma unroll
    for (int pr = 0; pr < 8; ++pr) {
      int hi = pr >> 2, g = pr & 3;
      int kk = kb + (pr >> 1) * 32 + half * 16 + (pr & 1) * 8;
      uint2 two = *(const uint2*)(xb + kk);
      A[2 * pr]     = (int)two.x;
      A[2 * pr + 1] = (int)two.y;
    }
    v16i B;
#pragma unroll
    for (int v = 0; v < 16; ++v) {
      int kk = kb + (v >> 2) * 32 + half * 16 + (v & 3) * 4;
      const float4 wv = *(const float4*)(wr + kk);
      B[v] = (int)pk4_e4m3(wv.x * inv_s, wv.y * inv_s, wv.z * inv_s, wv.w * inv_s);
    }
    acc = __builtin_amdgcn_wmma_f32_16x16x128_fp8_fp8(A, B, (short)0, acc, false, false);
  }
  if (lane < 16) part[(size_t)split * N + n0 + lane] = acc[0] * s;
}

__global__ void __launch_bounds__(256)
attn_kernel(const float* __restrict__ q,
            const float* __restrict__ knew,
            const float* __restrict__ vnew,
            const float* __restrict__ kc,
            const float* __restrict__ vc,
            float* __restrict__ out) {
  __shared__ float scores[REP][CUR_LEN];
  __shared__ float qs[REP * HD];
  __shared__ float red[REP][64];
  const int kvh = blockIdx.x;
  const int tid = threadIdx.x;
  { int h = tid >> 6; qs[tid] = q[(kvh * REP + h) * HD + (tid & 63)]; }
  __syncthreads();
  const float* kb = kc + (size_t)kvh * MAX_SEQ * HD;
  const float scl = 0.125f;
  for (int p = tid; p < CUR_LEN; p += 256) {
    const float* krow = (p == CACHE_LEN) ? (knew + kvh * HD) : (kb + (size_t)p * HD);
    float d0 = 0, d1 = 0, d2 = 0, d3 = 0;
    for (int t = 0; t < HD; t += 4) {
      float4 kv = *(const float4*)(krow + t);
      d0 += qs[0*HD+t]*kv.x + qs[0*HD+t+1]*kv.y + qs[0*HD+t+2]*kv.z + qs[0*HD+t+3]*kv.w;
      d1 += qs[1*HD+t]*kv.x + qs[1*HD+t+1]*kv.y + qs[1*HD+t+2]*kv.z + qs[1*HD+t+3]*kv.w;
      d2 += qs[2*HD+t]*kv.x + qs[2*HD+t+1]*kv.y + qs[2*HD+t+2]*kv.z + qs[2*HD+t+3]*kv.w;
      d3 += qs[3*HD+t]*kv.x + qs[3*HD+t+1]*kv.y + qs[3*HD+t+2]*kv.z + qs[3*HD+t+3]*kv.w;
    }
    scores[0][p] = d0 * scl; scores[1][p] = d1 * scl;
    scores[2][p] = d2 * scl; scores[3][p] = d3 * scl;
  }
  __syncthreads();
  const int h = tid >> 6, l = tid & 63;
  float mx = -1e30f;
  for (int p = l; p < CUR_LEN; p += 64) mx = fmaxf(mx, scores[h][p]);
  red[h][l] = mx; __syncthreads();
  for (int s = 32; s > 0; s >>= 1) { if (l < s) red[h][l] = fmaxf(red[h][l], red[h][l + s]); __syncthreads(); }
  mx = red[h][0]; __syncthreads();
  float sum = 0.0f;
  for (int p = l; p < CUR_LEN; p += 64) { float e = __expf(scores[h][p] - mx); scores[h][p] = e; sum += e; }
  red[h][l] = sum; __syncthreads();
  for (int s = 32; s > 0; s >>= 1) { if (l < s) red[h][l] += red[h][l + s]; __syncthreads(); }
  const float inv = 1.0f / red[h][0];
  __syncthreads();
  const float* vb = vc + (size_t)kvh * MAX_SEQ * HD;
  float acc = 0.0f;
  for (int p = 0; p < CACHE_LEN; ++p) acc += scores[h][p] * vb[(size_t)p * HD + l];
  acc += scores[h][CACHE_LEN] * vnew[kvh * HD + l];
  out[(kvh * REP + h) * HD + l] = acc * inv;
}

__global__ void __launch_bounds__(256)
gemv_f32_kernel(const float* __restrict__ W, const float* __restrict__ x,
                float* __restrict__ out, int K, int N) {
  int row  = blockIdx.x * 8 + (threadIdx.x >> 5);
  int lane = threadIdx.x & 31;
  if (row >= N) return;
  const float* wr = W + (size_t)row * K;
  float acc = 0.0f;
  for (int t = lane * 4; t < K; t += 128) {
    float4 wv = *(const float4*)(wr + t);
    float4 xv = *(const float4*)(x + t);
    acc += wv.x * xv.x + wv.y * xv.y + wv.z * xv.z + wv.w * xv.w;
  }
  for (int off = 16; off > 0; off >>= 1) acc += __shfl_down(acc, off, 32);
  if (lane == 0) out[row] = acc;
}

extern "C" void kernel_launch(void* const* d_in, const int* in_sizes, int n_in,
                              void* d_out, int out_size, void* d_ws, size_t ws_size,
                              hipStream_t stream) {
  (void)in_sizes; (void)n_in; (void)out_size; (void)ws_size;
  const int*   input_ids = (const int*)  d_in[0];
  const int*   pos_ids   = (const int*)  d_in[1];
  const float* embed_w   = (const float*)d_in[2];
  const float* wq  = (const float*)d_in[3];
  const float* wk  = (const float*)d_in[4];
  const float* wv  = (const float*)d_in[5];
  const float* wo  = (const float*)d_in[6];
  const float* wg  = (const float*)d_in[7];
  const float* wu  = (const float*)d_in[8];
  const float* wd  = (const float*)d_in[9];
  const float* ln1 = (const float*)d_in[10];
  const float* ln2 = (const float*)d_in[11];
  const float* norm_w  = (const float*)d_in[12];
  const float* lm_head = (const float*)d_in[13];
  const float* kc  = (const float*)d_in[14];
  const float* vc  = (const float*)d_in[15];
  float* logits = (float*)d_out;

  char* cur = (char*)d_ws;
  auto alloc = [&](size_t bytes) -> void* {
    void* r = (void*)cur; cur += (bytes + 255) & ~(size_t)255; return r;
  };
  unsigned int* amax    = (unsigned int*)alloc(28 * 4);
  float*        scales  = (float*)       alloc(28 * 4);
  float*        partial = (float*)       alloc((size_t)INTER * (D / KCHUNK) * 4);
  float*        h       = (float*)       alloc(D * 4);
  unsigned int* xq      = (unsigned int*)alloc(D);
  float*        q       = (float*)       alloc(D * 4);
  float*        k       = (float*)       alloc(KVH * HD * 4);
  float*        v       = (float*)       alloc(KVH * HD * 4);
  float*        qr      = (float*)       alloc(D * 4);
  float*        kr      = (float*)       alloc(KVH * HD * 4);
  float*        attn    = (float*)       alloc(D * 4);
  unsigned int* attq    = (unsigned int*)alloc(D);
  float*        proj    = (float*)       alloc(D * 4);
  float*        gate    = (float*)       alloc(INTER * 4);
  float*        up      = (float*)       alloc(INTER * 4);
  unsigned int* mlpq    = (unsigned int*)alloc(INTER);
  float*        mlpo    = (float*)       alloc(D * 4);
  float*        hfin    = (float*)       alloc(D * 4);

  auto fp8_gemv = [&](const float* W, const unsigned int* x, const float* sc,
                      float* outv, int K, int N) {
    int split = K / KCHUNK;
    dim3 g((unsigned)(N / 16), (unsigned)split);
    gemv_fp8_wmma_kernel<<<g, 32, 0, stream>>>(W, x, sc, partial, K, N);
    reduce_kernel<<<(N + 255) / 256, 256, 0, stream>>>(partial, outv, N, split);
  };

  zero_amax_kernel<<<1, 32, 0, stream>>>(amax);
  const float*  tbase[7] = { wq, wk, wv, wo, wg, wu, wd };
  const size_t  tsize[7] = { (size_t)D * D, (size_t)KVH * HD * D, (size_t)KVH * HD * D,
                             (size_t)D * D, (size_t)INTER * D, (size_t)INTER * D,
                             (size_t)D * INTER };
  for (int l = 0; l < N_LAYERS; ++l)
    for (int t = 0; t < 7; ++t) {
      int n4 = (int)(tsize[t] >> 2);
      int blocks = (n4 + 255) / 256; if (blocks > 2048) blocks = 2048;
      amax_kernel<<<blocks, 256, 0, stream>>>(tbase[t] + (size_t)l * tsize[t], n4,
                                              amax + l * 7 + t);
    }
  finalize_scales_kernel<<<1, 32, 0, stream>>>(amax, scales);

  embed_kernel<<<D / 256, 256, 0, stream>>>(embed_w, input_ids, h);

  for (int l = 0; l < N_LAYERS; ++l) {
    const float* sc = scales + l * 7;
    rmsnorm_kernel<<<1, 256, 0, stream>>>(h, ln1 + (size_t)l * D, (float*)nullptr, xq, D);
    fp8_gemv(wq + (size_t)l * D * D,         xq, sc + 0, q, D, D);
    fp8_gemv(wk + (size_t)l * KVH * HD * D,  xq, sc + 1, k, D, KVH * HD);
    fp8_gemv(wv + (size_t)l * KVH * HD * D,  xq, sc + 2, v, D, KVH * HD);
    rope_kernel<<<((H + KVH) * HD) / 256, 256, 0, stream>>>(q, k, pos_ids, qr, kr);
    attn_kernel<<<KVH, 256, 0, stream>>>(qr, kr, v,
        kc + (size_t)l * KVH * MAX_SEQ * HD, vc + (size_t)l * KVH * MAX_SEQ * HD, attn);
    quant_kernel<<<(D / 4) / 256, 256, 0, stream>>>(attn, attq, D / 4);
    fp8_gemv(wo + (size_t)l * D * D, attq, sc + 3, proj, D, D);
    add_kernel<<<D / 256, 256, 0, stream>>>(h, proj, D);

    rmsnorm_kernel<<<1, 256, 0, stream>>>(h, ln2 + (size_t)l * D, (float*)nullptr, xq, D);
    fp8_gemv(wg + (size_t)l * INTER * D, xq, sc + 4, gate, D, INTER);
    fp8_gemv(wu + (size_t)l * INTER * D, xq, sc + 5, up,   D, INTER);
    silu_mul_q_kernel<<<(INTER / 4) / 256, 256, 0, stream>>>(gate, up, mlpq, INTER / 4);
    fp8_gemv(wd + (size_t)l * D * INTER, mlpq, sc + 6, mlpo, INTER, D);
    add_kernel<<<D / 256, 256, 0, stream>>>(h, mlpo, D);
  }

  rmsnorm_kernel<<<1, 256, 0, stream>>>(h, norm_w, hfin, (unsigned int*)nullptr, D);
  gemv_f32_kernel<<<VOCAB / 8, 256, 0, stream>>>(lm_head, hfin, logits, D, VOCAB);
}
